// AnswerDecoder_65730179498176
// MI455X (gfx1250) — matrix operand order repaired and checked by reading, not compiled, not run on hardware
//
#include <hip/hip_runtime.h>

// ---------------------------------------------------------------------------
// Types for gfx1250 WMMA (wave32): v16bf A/B fragments, v8f accumulator.
// ---------------------------------------------------------------------------
typedef __attribute__((ext_vector_type(16))) __bf16 bf16x16;
typedef __attribute__((ext_vector_type(8)))  __bf16 bf16x8;
typedef __attribute__((ext_vector_type(8)))  float  floatx8;

union ABFrag { bf16x16 v; bf16x8 h[2]; };

#define B_   64
#define T_   32
#define R_   49
#define LOCAL_ 1024
#define QVEC_  512
#define EMB_   256
#define HID_   512
#define VOCAB_ 10000
#define KCAT_  1280   // EMB + HID (x part) + HID (h part)

// ---------------------------------------------------------------------------
// Generic bf16 WMMA GEMM:  C[M,N] (f32) = A[M,K](bf16) * W[N,K](bf16)^T
// (torch-style weight layout: both A and W are K-contiguous).
// Each wave computes 1 M-tile x NT N-tiles (16x(16*NT) of C); the A fragment
// is loaded once per K-step and reused for NT wmmas. 8 waves / block.
// act: 0 = none, 1 = tanh.  Optional bias[N].  Optional bf16 secondary output.
// Requires: M%16==0, N%(16*NT)==0, K%32==0.
// ---------------------------------------------------------------------------
template <int NT>
__global__ __launch_bounds__(256) void wmma_gemm_bf16(
    const __bf16* __restrict__ A, int lda,
    const __bf16* __restrict__ W, int ldw,
    float* __restrict__ C, int ldc,
    int M, int N, int K,
    const float* __restrict__ bias, int act,
    __bf16* __restrict__ Cbf, int ldcbf)
{
    const int wave = threadIdx.x >> 5;
    const int lane = threadIdx.x & 31;
    const int ngrp = N / (16 * NT);                 // N-tile groups
    const long total = (long)(M >> 4) * (long)ngrp;
    const long tile  = (long)blockIdx.x * 8 + wave;
    if (tile >= total) return;                      // wave-uniform exit

    const int tm = (int)(tile / ngrp);
    const int tg = (int)(tile % ngrp);
    const int m0 = tm << 4;
    const int n0 = tg * 16 * NT;

    const int g = lane >> 4;                        // lane group 0/1
    const int r = lane & 15;

    const __bf16* Arow = A + (long)(m0 + r) * lda;  // A fragment row (m = m0+r)
    const __bf16* Wrow[NT];
    #pragma unroll
    for (int j = 0; j < NT; ++j)                    // B fragment col (n = n0+16j+r)
        Wrow[j] = W + (long)(n0 + 16 * j + r) * ldw;

    floatx8 acc[NT];
    #pragma unroll
    for (int j = 0; j < NT; ++j) acc[j] = (floatx8){};

    #pragma unroll 2
    for (int k0 = 0; k0 < K; k0 += 32) {
        // A 16x32 bf16 fragment: lane holds k in [16g, 16g+15]
        ABFrag af;
        af.h[0] = *(const bf16x8*)(Arow + k0 + 16 * g);
        af.h[1] = *(const bf16x8*)(Arow + k0 + 16 * g + 8);

        if (k0 + 32 < K) {                          // WGP-scope prefetch of next K slice
            __builtin_prefetch(Arow + k0 + 32, 0, 3);
            __builtin_prefetch(Wrow[0] + k0 + 32, 0, 3);
            __builtin_prefetch(Wrow[NT - 1] + k0 + 32, 0, 3);
        }
        #pragma unroll
        for (int j = 0; j < NT; ++j) {
            // B 32x16 bf16 fragment: lane holds k in [16g, 16g+15] for its column
            bf16x16 bf = *(const bf16x16*)(Wrow[j] + k0 + 16 * g);
            acc[j] = __builtin_amdgcn_wmma_f32_16x16x32_bf16(
                         false, af.v, false, bf, (short)0, acc[j], false, false);
        }
    }

    // C/D layout: lane writes column n, rows m0 + 8g + i
    #pragma unroll
    for (int j = 0; j < NT; ++j) {
        const int n = n0 + 16 * j + r;
        float bv = bias ? bias[n] : 0.0f;
        #pragma unroll
        for (int i = 0; i < 8; ++i) {
            const int m = m0 + 8 * g + i;
            float v = acc[j][i] + bv;
            if (act == 1) v = tanhf(v);
            C[(long)m * ldc + n] = v;
            if (Cbf) Cbf[(long)m * ldcbf + n] = (__bf16)v;
        }
    }
}

// ---------------------------------------------------------------------------
// One-time helpers
// ---------------------------------------------------------------------------
__global__ __launch_bounds__(256) void to_bf16_k(const float* __restrict__ s,
                                                 __bf16* __restrict__ d, long n)
{
    long i = (long)blockIdx.x * 256 + threadIdx.x;
    if (i < n) d[i] = (__bf16)s[i];
}

// Wfull[2048 x 1280] = [W_ih | W_hh] (bf16), so gates = xh @ Wfull^T
__global__ __launch_bounds__(256) void build_wfull_k(const float* __restrict__ wih,
                                                     const float* __restrict__ whh,
                                                     __bf16* __restrict__ wf)
{
    long idx = (long)blockIdx.x * 256 + threadIdx.x;
    if (idx >= (long)4 * HID_ * KCAT_) return;
    int nrow = (int)(idx / KCAT_);
    int k    = (int)(idx % KCAT_);
    float v  = (k < EMB_ + HID_) ? wih[(long)nrow * (EMB_ + HID_) + k]
                                 : whh[(long)nrow * HID_ + (k - EMB_ - HID_)];
    wf[idx] = (__bf16)v;
}

__global__ __launch_bounds__(256) void vec_add_k(const float* __restrict__ a,
                                                 const float* __restrict__ b,
                                                 float* __restrict__ o, int n)
{
    int i = blockIdx.x * 256 + threadIdx.x;
    if (i < n) o[i] = a[i] + b[i];
}

// ---------------------------------------------------------------------------
// Per-step: xh = [emb(y_t) | o_prev | h_prev]  (bf16, [64 x 1280])
// ---------------------------------------------------------------------------
__global__ __launch_bounds__(256) void prep_xh_k(const __bf16* __restrict__ emb_bf,
                                                 const int* __restrict__ answers,
                                                 const float* __restrict__ o_st,
                                                 const float* __restrict__ h_st,
                                                 __bf16* __restrict__ xh, int t)
{
    int idx = blockIdx.x * 256 + threadIdx.x;
    if (idx >= B_ * KCAT_) return;
    int b = idx / KCAT_, c = idx % KCAT_;
    __bf16 v;
    if (c < EMB_) {
        int tok = (t == 0) ? 1 : answers[b * T_ + t - 1];   // START_IDX = 1
        v = emb_bf[(long)tok * EMB_ + c];
    } else if (c < EMB_ + HID_) {
        v = (__bf16)o_st[b * HID_ + (c - EMB_)];
    } else {
        v = (__bf16)h_st[b * HID_ + (c - EMB_ - HID_)];
    }
    xh[idx] = v;
}

// ---------------------------------------------------------------------------
// LSTM gate nonlinearity: gates [64 x 2048] (i|f|g|o) -> update c,h (f32)
// ---------------------------------------------------------------------------
__global__ __launch_bounds__(256) void lstm_act_k(const float* __restrict__ gates,
                                                  float* __restrict__ c_st,
                                                  float* __restrict__ h_st)
{
    int idx = blockIdx.x * 256 + threadIdx.x;
    if (idx >= B_ * HID_) return;
    int b = idx >> 9, j = idx & (HID_ - 1);
    const float* gr = gates + (long)b * 4 * HID_;
    float ig = gr[j], fg = gr[HID_ + j], gg = gr[2 * HID_ + j], og = gr[3 * HID_ + j];
    float si = 1.0f / (1.0f + __expf(-ig));
    float sf = 1.0f / (1.0f + __expf(-fg));
    float so = 1.0f / (1.0f + __expf(-og));
    float c  = sf * c_st[idx] + si * tanhf(gg);
    c_st[idx] = c;
    h_st[idx] = so * tanhf(c);
}

// ---------------------------------------------------------------------------
// Attention per batch row: e = attn_proj . h ; softmax ; a = local^T att ;
// emits u = [a | h] as bf16 for the o-GEMM.  One 256-thread block per b.
// ---------------------------------------------------------------------------
__global__ __launch_bounds__(256) void attn_k(const float* __restrict__ attn_proj, // [B,R,H]
                                              const float* __restrict__ local,     // [B,R,F]
                                              const float* __restrict__ h_st,      // [B,H]
                                              __bf16* __restrict__ u)              // [B, F+H]
{
    const int b = blockIdx.x;
    __shared__ float hs[HID_];
    __shared__ float ev[64];
    __shared__ float att[64];

    for (int i = threadIdx.x; i < HID_; i += 256) hs[i] = h_st[b * HID_ + i];
    __syncthreads();

    const int wave = threadIdx.x >> 5, lane = threadIdx.x & 31;
    for (int r = wave; r < R_; r += 8) {
        const float* ap = attn_proj + ((long)b * R_ + r) * HID_;
        float s = 0.0f;
        for (int j = lane; j < HID_; j += 32) s += ap[j] * hs[j];
        for (int off = 16; off; off >>= 1) s += __shfl_xor(s, off, 32);
        if (lane == 0) ev[r] = s;
    }
    __syncthreads();

    if (threadIdx.x == 0) {
        float mx = ev[0];
        for (int r = 1; r < R_; ++r) mx = fmaxf(mx, ev[r]);
        float sum = 0.0f;
        for (int r = 0; r < R_; ++r) { float v = __expf(ev[r] - mx); att[r] = v; sum += v; }
        float inv = 1.0f / sum;
        for (int r = 0; r < R_; ++r) att[r] *= inv;
    }
    __syncthreads();

    const float* lb = local + (long)b * R_ * LOCAL_;
    for (int f = threadIdx.x; f < LOCAL_; f += 256) {
        float s = 0.0f;
        #pragma unroll 7
        for (int r = 0; r < R_; ++r) s += lb[r * LOCAL_ + f] * att[r];
        u[(long)b * (LOCAL_ + HID_) + f] = (__bf16)s;
    }
    for (int j = threadIdx.x; j < HID_; j += 256)
        u[(long)b * (LOCAL_ + HID_) + LOCAL_ + j] = (__bf16)hs[j];
}

// ---------------------------------------------------------------------------
// Host side
// ---------------------------------------------------------------------------
static inline char* carve(char*& p, size_t bytes)
{
    char* r = p;
    p += (bytes + 255) & ~(size_t)255;
    return r;
}

static inline void launch_gemm(hipStream_t s,
                               const __bf16* A, int lda, const __bf16* W, int ldw,
                               float* C, int ldc, int M, int N, int K,
                               const float* bias, int act, __bf16* Cbf, int ldcbf)
{
    // Pick the widest register-blocking the N dimension allows.
    if (N % 80 == 0 && N % 64 != 0) {               // e.g. N = 10000 = 80*125
        long tiles  = (long)(M >> 4) * (long)(N / 80);
        int  blocks = (int)((tiles + 7) / 8);
        wmma_gemm_bf16<5><<<blocks, 256, 0, s>>>(A, lda, W, ldw, C, ldc, M, N, K,
                                                 bias, act, Cbf, ldcbf);
    } else if (N % 64 == 0) {                       // e.g. N = 512, 2048
        long tiles  = (long)(M >> 4) * (long)(N / 64);
        int  blocks = (int)((tiles + 7) / 8);
        wmma_gemm_bf16<4><<<blocks, 256, 0, s>>>(A, lda, W, ldw, C, ldc, M, N, K,
                                                 bias, act, Cbf, ldcbf);
    } else {                                        // generic fallback
        long tiles  = (long)(M >> 4) * (long)(N >> 4);
        int  blocks = (int)((tiles + 7) / 8);
        wmma_gemm_bf16<1><<<blocks, 256, 0, s>>>(A, lda, W, ldw, C, ldc, M, N, K,
                                                 bias, act, Cbf, ldcbf);
    }
}

extern "C" void kernel_launch(void* const* d_in, const int* in_sizes, int n_in,
                              void* d_out, int out_size, void* d_ws, size_t ws_size,
                              hipStream_t stream)
{
    (void)in_sizes; (void)n_in; (void)out_size; (void)ws_size;

    const float* local_f   = (const float*)d_in[0];   // [B,R,LOCAL]
    const float* global_f  = (const float*)d_in[1];   // [B,2*LOCAL]
    const float* qvec      = (const float*)d_in[2];   // [B,QVEC]
    const int*   answers   = (const int*)  d_in[3];   // [B,T]
    const float* emb       = (const float*)d_in[4];   // [VOCAB,EMB]
    const float* W_g2o     = (const float*)d_in[5];   // [HID,2*LOCAL]
    const float* b_g2o     = (const float*)d_in[6];
    const float* W_h       = (const float*)d_in[7];   // [HID,QVEC]
    const float* W_c       = (const float*)d_in[8];
    const float* W_ih      = (const float*)d_in[9];   // [4H, EMB+HID]
    const float* W_hh      = (const float*)d_in[10];  // [4H, HID]
    const float* b_ih      = (const float*)d_in[11];
    const float* b_hh      = (const float*)d_in[12];
    const float* W_attn    = (const float*)d_in[13];  // [HID, LOCAL]
    const float* W_u       = (const float*)d_in[14];  // [HID, LOCAL+HID]
    const float* b_u       = (const float*)d_in[15];
    const float* W_vocab   = (const float*)d_in[16];  // [VOCAB, HID]
    const float* b_vocab   = (const float*)d_in[17];
    float* out             = (float*)d_out;           // [B,T,VOCAB]

    // ---- workspace carve --------------------------------------------------
    char* p = (char*)d_ws;
    __bf16* emb_bf   = (__bf16*)carve(p, (size_t)VOCAB_ * EMB_ * 2);
    __bf16* local_bf = (__bf16*)carve(p, (size_t)B_ * R_ * LOCAL_ * 2);
    __bf16* q_bf     = (__bf16*)carve(p, (size_t)B_ * QVEC_ * 2);
    __bf16* g_bf     = (__bf16*)carve(p, (size_t)B_ * 2 * LOCAL_ * 2);
    __bf16* Wg2o_bf  = (__bf16*)carve(p, (size_t)HID_ * 2 * LOCAL_ * 2);
    __bf16* Wh_bf    = (__bf16*)carve(p, (size_t)HID_ * QVEC_ * 2);
    __bf16* Wc_bf    = (__bf16*)carve(p, (size_t)HID_ * QVEC_ * 2);
    __bf16* Wattn_bf = (__bf16*)carve(p, (size_t)HID_ * LOCAL_ * 2);
    __bf16* Wu_bf    = (__bf16*)carve(p, (size_t)HID_ * (LOCAL_ + HID_) * 2);
    __bf16* Wvoc_bf  = (__bf16*)carve(p, (size_t)VOCAB_ * HID_ * 2);
    __bf16* Wfull_bf = (__bf16*)carve(p, (size_t)4 * HID_ * KCAT_ * 2);
    float*  bsum     = (float*) carve(p, (size_t)4 * HID_ * 4);
    float*  attn_pr  = (float*) carve(p, (size_t)B_ * R_ * HID_ * 4);
    float*  h_st     = (float*) carve(p, (size_t)B_ * HID_ * 4);
    float*  c_st     = (float*) carve(p, (size_t)B_ * HID_ * 4);
    float*  o_st     = (float*) carve(p, (size_t)B_ * HID_ * 4);
    __bf16* xh_bf    = (__bf16*)carve(p, (size_t)B_ * KCAT_ * 2);
    float*  gates    = (float*) carve(p, (size_t)B_ * 4 * HID_ * 4);
    __bf16* u_bf     = (__bf16*)carve(p, (size_t)B_ * (LOCAL_ + HID_) * 2);
    __bf16* O_all    = (__bf16*)carve(p, (size_t)B_ * T_ * HID_ * 2);   // row = b*T + t

    auto cvt = [&](const float* src, __bf16* dst, long n) {
        to_bf16_k<<<(int)((n + 255) / 256), 256, 0, stream>>>(src, dst, n);
    };

    // ---- one-time conversions / fusions ----------------------------------
    cvt(emb,      emb_bf,   (long)VOCAB_ * EMB_);
    cvt(local_f,  local_bf, (long)B_ * R_ * LOCAL_);
    cvt(qvec,     q_bf,     (long)B_ * QVEC_);
    cvt(global_f, g_bf,     (long)B_ * 2 * LOCAL_);
    cvt(W_g2o,    Wg2o_bf,  (long)HID_ * 2 * LOCAL_);
    cvt(W_h,      Wh_bf,    (long)HID_ * QVEC_);
    cvt(W_c,      Wc_bf,    (long)HID_ * QVEC_);
    cvt(W_attn,   Wattn_bf, (long)HID_ * LOCAL_);
    cvt(W_u,      Wu_bf,    (long)HID_ * (LOCAL_ + HID_));
    cvt(W_vocab,  Wvoc_bf,  (long)VOCAB_ * HID_);
    {
        long n = (long)4 * HID_ * KCAT_;
        build_wfull_k<<<(int)((n + 255) / 256), 256, 0, stream>>>(W_ih, W_hh, Wfull_bf);
        vec_add_k<<<(4 * HID_ + 255) / 256, 256, 0, stream>>>(b_ih, b_hh, bsum, 4 * HID_);
    }

    // ---- initial states (WMMA GEMMs) -------------------------------------
    // h0 = q @ W_h^T ; c0 = q @ W_c^T ; o0 = g @ W_g2o^T + b_g2o
    launch_gemm(stream, q_bf, QVEC_,      Wh_bf,   QVEC_,      h_st, HID_, B_, HID_, QVEC_,      nullptr, 0, nullptr, 0);
    launch_gemm(stream, q_bf, QVEC_,      Wc_bf,   QVEC_,      c_st, HID_, B_, HID_, QVEC_,      nullptr, 0, nullptr, 0);
    launch_gemm(stream, g_bf, 2 * LOCAL_, Wg2o_bf, 2 * LOCAL_, o_st, HID_, B_, HID_, 2 * LOCAL_, b_g2o,   0, nullptr, 0);
    // attn_proj[B*R, HID] = local @ W_attn^T   (loop-invariant, hoisted)
    launch_gemm(stream, local_bf, LOCAL_, Wattn_bf, LOCAL_, attn_pr, HID_,
                B_ * R_, HID_, LOCAL_, nullptr, 0, nullptr, 0);

    // ---- sequential recurrence (T steps) ---------------------------------
    for (int t = 0; t < T_; ++t) {
        prep_xh_k<<<(B_ * KCAT_ + 255) / 256, 256, 0, stream>>>(
            emb_bf, answers, o_st, h_st, xh_bf, t);
        // gates = xh @ Wfull^T + (b_ih + b_hh)
        launch_gemm(stream, xh_bf, KCAT_, Wfull_bf, KCAT_, gates, 4 * HID_,
                    B_, 4 * HID_, KCAT_, bsum, 0, nullptr, 0);
        lstm_act_k<<<(B_ * HID_ + 255) / 256, 256, 0, stream>>>(gates, c_st, h_st);
        attn_k<<<B_, 256, 0, stream>>>(attn_pr, local_f, h_st, u_bf);
        // o = tanh(u @ W_u^T + b_u) -> o_st (f32, next-step state)
        //   and bf16 copy into O_all at row b*T + t (ready for the big GEMM)
        launch_gemm(stream, u_bf, LOCAL_ + HID_, Wu_bf, LOCAL_ + HID_, o_st, HID_,
                    B_, HID_, LOCAL_ + HID_, b_u, 1, O_all + (long)t * HID_, T_ * HID_);
    }

    // ---- hoisted vocab projection: one [2048 x 512] x [512 x 10000] GEMM --
    // O_all rows are (b*T + t) so the output is directly [B,T,VOCAB] row-major.
    launch_gemm(stream, O_all, HID_, Wvoc_bf, HID_, out, VOCAB_,
                B_ * T_, VOCAB_, HID_, b_vocab, 0, nullptr, 0);
}
